// WindowAttention_4561255268505
// MI455X (gfx1250) — matrix operand order repaired and checked by reading, not compiled, not run on hardware
//
#include <hip/hip_runtime.h>

#define WSZ   8
#define NTOK  64
#define NHEAD 8
#define HD    16
#define DIM   128
#define HI    256
#define WID   256

typedef _Float16 h16;
typedef __attribute__((ext_vector_type(8)))  h16   v8h;
typedef __attribute__((ext_vector_type(16))) h16   v16h;
typedef __attribute__((ext_vector_type(8)))  float v8f;
typedef __attribute__((ext_vector_type(4)))  float v4f;

// LDS layout (bytes)
#define X_OFF   0          // x tile / ctx  : 64 x 136 halfs = 17408 B
#define Q_OFF   17408      // q[8][64][16]  : 16384 B
#define K_OFF   33792      // k[8][64][16]  : 16384 B   (= Q_OFF + 16384)
#define V_OFF   50176      // v[8][16][64]  : 16384 B   (= Q_OFF + 32768, transposed)
#define S_OFF   66560      // scores[4][64][64] f32 : 65536 B (probs f16 aliased in-row)
#define I_OFF   132096     // invsum[4][64] f32 : 1024 B
#define LDS_BYTES 133120

__device__ __forceinline__ v16h join8(v8h lo, v8h hi) {
  return __builtin_shufflevector(lo, hi, 0,1,2,3,4,5,6,7,8,9,10,11,12,13,14,15);
}
__device__ __forceinline__ v8h ld8(const h16* p) { return *(const v8h*)p; }

__device__ __forceinline__ v8f wmma16(v16h a, v16h b, v8f c) {
  // D = A(16x32 f16) * B(32x16 f16) + C(16x16 f32)
  return __builtin_amdgcn_wmma_f32_16x16x32_f16(false, a, false, b, (short)0, c,
                                                false, false);
}

// A fragment from row-major f16 source: lane l -> row row0+(l&15),
// chunks K = kbase + (l>=16?8:0)+[0..7] and +16..+23.
__device__ __forceinline__ v16h load_a(const h16* base, int row0, int stride,
                                       int kbase, int lane) {
  const h16* p = base + (row0 + (lane & 15)) * stride + kbase + ((lane >> 4) << 3);
  return join8(ld8(p), ld8(p + 16));
}

// B fragment from N-major (row = N, K contiguous) f16 source:
// lane l -> col n (caller passes n0 + (l&15)), K = kbase + (l>=16?8:0)+[0..7] and +16..+23.
__device__ __forceinline__ v16h load_b_nk(const h16* base, int n, int stride,
                                          int kbase, int lane) {
  const h16* p = base + n * stride + kbase + ((lane >> 4) << 3);
  return join8(ld8(p), ld8(p + 16));
}

__global__ void conv_weights(const float* __restrict__ qkvw,
                             const float* __restrict__ projw,
                             h16* __restrict__ qkvw_h, h16* __restrict__ projw_h) {
  int i = blockIdx.x * 256 + threadIdx.x;
  if (i < 3 * DIM * DIM) qkvw_h[i] = (h16)qkvw[i];
  if (i < DIM * DIM)     projw_h[i] = (h16)projw[i];
}

__global__ void build_bias(const float* __restrict__ rel_bias,
                           const int* __restrict__ rel_index,
                           float* __restrict__ bias8) {
  int i = blockIdx.x * 256 + threadIdx.x;           // 8 * 4096
  if (i < NHEAD * NTOK * NTOK) {
    int h = i >> 12, ij = i & 4095;
    bias8[i] = rel_bias[rel_index[ij] * NHEAD + h];
  }
}

__global__ void __launch_bounds__(256)
win_attn(const float* __restrict__ x, const float* __restrict__ qkv_b,
         const float* __restrict__ proj_b, const h16* __restrict__ qkvw_h,
         const h16* __restrict__ projw_h, const float* __restrict__ bias8,
         float* __restrict__ out) {
  extern __shared__ char smem[];
  h16*   xctx = (h16*)(smem + X_OFF);     // [64][136] x tile, later ctx
  h16*   qlds = (h16*)(smem + Q_OFF);     // [head][tok][16]
  h16*   klds = (h16*)(smem + K_OFF);     // [head][tok][16]
  h16*   vlds = (h16*)(smem + V_OFF);     // [head][dim][64]
  float* sco  = (float*)(smem + S_OFF);   // [hl][row][64]
  float* inv  = (float*)(smem + I_OFF);   // [hl][64]

  const int tid  = threadIdx.x;
  const int wave = tid >> 5, lane = tid & 31;
  const int m  = lane & 15;
  const int hi = lane >> 4;
  const int blk = blockIdx.x;
  const int bb  = blk >> 10;
  const int whi = (blk >> 5) & 31;
  const int wwi = blk & 31;
  const float* xwin = x + (size_t)bb * DIM * HI * WID + whi * 8 * WID + wwi * 8;

  // ---- Stage 1: x window -> LDS f16 (float4 loads) ----
#pragma unroll
  for (int it = 0; it < 8; ++it) {
    int idx = it * 256 + tid;            // 0..2047 chunks of 4 floats
    int c = idx >> 4;
    int rem = idx & 15;
    int i = rem >> 1, jh = rem & 1;
    v4f v = *(const v4f*)(xwin + (size_t)c * HI * WID + i * WID + jh * 4);
    int tokb = i * 8 + jh * 4;
#pragma unroll
    for (int q = 0; q < 4; ++q) xctx[(tokb + q) * 136 + c] = (h16)v[q];
  }
  __syncthreads();

  // ---- Stage 2: QKV GEMM (64x128)x(128x384), write q,k,v^T ----
  for (int cc = wave * 3; cc < wave * 3 + 3; ++cc) {
    float bq = qkv_b[cc * 16 + m];
    // q/k/v LDS regions are contiguous: branchless destination base.
    h16* dstbase = qlds + (cc >> 3) * 8192 + (cc & 7) * 1024;  // halfs
    const bool isv = (cc >= 16);
#pragma unroll
    for (int r = 0; r < 4; ++r) {
      v8f acc = {};
#pragma unroll
      for (int kb = 0; kb < 4; ++kb) {
        v16h a = load_a(xctx, r * 16, 136, kb * 32, lane);
        v16h b = load_b_nk(qkvw_h, cc * 16 + m, DIM, kb * 32, lane);
        acc = wmma16(a, b, acc);
      }
#pragma unroll
      for (int vi = 0; vi < 8; ++vi) {
        int tok = r * 16 + vi + hi * 8;
        int idx = isv ? (m * 64 + tok) : (tok * 16 + m);  // v transposed
        dstbase[idx] = (h16)(acc[vi] + bq);
      }
    }
  }
  __syncthreads();

  const float scale = 0.25f;  // hd^-0.5
  const int hl = wave >> 1;   // head slot within pass
  const int rh = wave & 1;    // row half

  for (int pass = 0; pass < 2; ++pass) {
    const int head = pass * 4 + hl;
    // ---- Stage 3: scores = q @ k^T  (K=16 zero-padded to 32) ----
#pragma unroll
    for (int r = 2 * rh; r < 2 * rh + 2; ++r) {
#pragma unroll
      for (int cc = 0; cc < 4; ++cc) {
        v8h z8 = {};
        v16h a = join8(ld8(qlds + head * 1024 + (r * 16 + m) * 16 + hi * 8), z8);
        v16h b = join8(ld8(klds + head * 1024 + (cc * 16 + m) * 16 + hi * 8), z8);
        v8f acc = {};
        acc = wmma16(a, b, acc);
#pragma unroll
        for (int vi = 0; vi < 8; ++vi)
          sco[hl * 4096 + (r * 16 + vi + hi * 8) * 64 + cc * 16 + m] = acc[vi];
      }
    }
    __syncthreads();

    // ---- softmax: 256 threads = 4 heads x 64 rows; probs (unnormalized f16)
    //      written in-place at row start; 1/sum folded into attn@v epilogue.
    {
      int hl2 = tid >> 6, row = tid & 63;
      float* srow = sco + hl2 * 4096 + row * 64;
      const float* brow = bias8 + (pass * 4 + hl2) * 4096 + row * 64;
      float mx = -1e30f;
#pragma unroll 8
      for (int j = 0; j < 64; ++j) mx = fmaxf(mx, srow[j] * scale + brow[j]);
      float sum = 0.f;
      h16* prow = (h16*)srow;
#pragma unroll 8
      for (int j = 0; j < 64; ++j) {
        float e = __expf(srow[j] * scale + brow[j] - mx);
        sum += e;
        prow[j] = (h16)e;      // bytes 2j <= 4j: safe same-thread alias
      }
      inv[hl2 * 64 + row] = 1.0f / sum;
    }
    __syncthreads();

    // ---- Stage 4: ctx = probs @ v  (K=64, 2 steps) ----
#pragma unroll
    for (int r = 2 * rh; r < 2 * rh + 2; ++r) {
      v8f acc = {};
#pragma unroll
      for (int kb = 0; kb < 2; ++kb) {
        const h16* pa = (const h16*)(sco + hl * 4096) + (r * 16 + m) * 128 +
                        kb * 32 + hi * 8;
        v16h a = join8(ld8(pa), ld8(pa + 16));
        const h16* pb = vlds + head * 1024 + m * 64 + kb * 32 + hi * 8;
        v16h b = join8(ld8(pb), ld8(pb + 16));
        acc = wmma16(a, b, acc);
      }
#pragma unroll
      for (int vi = 0; vi < 8; ++vi) {
        int tok = r * 16 + vi + hi * 8;
        xctx[tok * 136 + head * 16 + m] = (h16)(acc[vi] * inv[hl * 64 + tok]);
      }
    }
    __syncthreads();
  }

  // ---- Stage 5: proj GEMM (64x128)x(128x128) -> stage in LDS ----
  float* pout = sco;  // reuse scores region: [o][64] f32 = 32 KB
  {
    const int cc = wave;
    float bo = proj_b[cc * 16 + m];
#pragma unroll
    for (int r = 0; r < 4; ++r) {
      v8f acc = {};
#pragma unroll
      for (int kb = 0; kb < 4; ++kb) {
        v16h a = load_a(xctx, r * 16, 136, kb * 32, lane);
        v16h b = load_b_nk(projw_h, cc * 16 + m, DIM, kb * 32, lane);
        acc = wmma16(a, b, acc);
      }
#pragma unroll
      for (int vi = 0; vi < 8; ++vi) {
        int tok = r * 16 + vi + hi * 8;
        pout[(cc * 16 + m) * 64 + tok] = acc[vi] + bo;
      }
    }
  }
  __syncthreads();

  // ---- Stage 6: coalesced f32 output stores (two float4 per (o,i) row) ----
  float* obase = out + ((size_t)bb * DIM * HI + (size_t)whi * 8) * WID + wwi * 8;
#pragma unroll
  for (int it = 0; it < 4; ++it) {
    int q = it * 256 + tid;          // 0..1023 : o = q>>3, i = q&7
    int o = q >> 3, i = q & 7;
    const float* src = pout + o * 64 + i * 8;
    float* dst = obase + (size_t)o * HI * WID + i * WID;
    *(v4f*)dst       = *(const v4f*)src;
    *(v4f*)(dst + 4) = *(const v4f*)(src + 4);
  }
}

extern "C" void kernel_launch(void* const* d_in, const int* in_sizes, int n_in,
                              void* d_out, int out_size, void* d_ws, size_t ws_size,
                              hipStream_t stream) {
  const float* x        = (const float*)d_in[0];
  const float* qkv_w    = (const float*)d_in[1];
  const float* qkv_b    = (const float*)d_in[2];
  const float* proj_w   = (const float*)d_in[3];
  const float* proj_b   = (const float*)d_in[4];
  const float* rel_bias = (const float*)d_in[5];
  const int*   rel_idx  = (const int*)d_in[6];
  float* out = (float*)d_out;

  h16*   qkvw_h  = (h16*)d_ws;                       // 98304 B
  h16*   projw_h = (h16*)((char*)d_ws + 98304);      // 32768 B
  float* bias8   = (float*)((char*)d_ws + 131072);   // 131072 B

  conv_weights<<<192, 256, 0, stream>>>(qkv_w, proj_w, qkvw_h, projw_h);
  build_bias<<<128, 256, 0, stream>>>(rel_bias, rel_idx, bias8);

  (void)hipFuncSetAttribute(reinterpret_cast<const void*>(win_attn),
                            hipFuncAttributeMaxDynamicSharedMemorySize, LDS_BYTES);

  win_attn<<<2048, 256, LDS_BYTES, stream>>>(x, qkv_b, proj_b, qkvw_h, projw_h,
                                             bias8, out);
}
